// SelfAttention_63075889709213
// MI455X (gfx1250) — matrix operand order repaired and checked by reading, not compiled, not run on hardware
//
#include <hip/hip_runtime.h>
#include <math.h>

typedef _Float16 half_t;
typedef _Float16 v16h __attribute__((ext_vector_type(16)));
typedef float    v8f  __attribute__((ext_vector_type(8)));

union Frag { v16h h; uint4 q[2]; };

static __device__ __forceinline__ v8f wmma_f16(v16h a, v16h b, v8f c) {
  // D = A(16x32 f16) * B(32x16 f16) + C(16x16 f32)
  return __builtin_amdgcn_wmma_f32_16x16x32_f16(false, a, false, b, (short)0, c,
                                                false, false);
}

// CDNA5 async copy: global -> LDS, 16B per lane, tracked on ASYNCcnt.
static __device__ __forceinline__ void async_copy_b128(unsigned lds_byte_off,
                                                       const void* gaddr) {
  asm volatile("global_load_async_to_lds_b128 %0, %1, off"
               :: "v"(lds_byte_off), "v"(gaddr) : "memory");
}
static __device__ __forceinline__ void wait_async0() {
  asm volatile("s_wait_asynccnt 0x0" ::: "memory");
}
// async loads complete in order: <=4 outstanding means all older ones landed
static __device__ __forceinline__ void wait_async4() {
  asm volatile("s_wait_asynccnt 0x4" ::: "memory");
}
static __device__ __forceinline__ void wait_ds() {
  asm volatile("s_wait_dscnt 0x0" ::: "memory");
}
// low 32 bits of a flat LDS address == workgroup LDS byte offset
static __device__ __forceinline__ unsigned lds_off(const void* p) {
  return (unsigned)(size_t)p;
}

// ---------------------------------------------------------------------------
// Tiled GEMM: C[M,N] = A[M,K] * B[K,N]
// A: fp32 (AH=false, register-pipelined + cvt) or f16 (AH=true, async->LDS).
// B: fp32, register-pipelined, converted + transposed into Bs[n][k].
// C modes: 0 = fp32, 1 = f16, 2 = f16 transposed (C[n][m], packed b128 store)
// Block 128x128, K-step 64 (2 WMMA K-steps per stage), 256 thr = 8 waves.
// ---------------------------------------------------------------------------
#define BM 128
#define BN 128
#define BK 64

enum { C_F32 = 0, C_F16 = 1, C_F16T = 2 };

template <bool AH, int CM>
__global__ __launch_bounds__(256) void gemm_kernel(const void* __restrict__ Av,
                                                   const float* __restrict__ B,
                                                   void* __restrict__ Cv,
                                                   int M, int N, int K) {
  __shared__ half_t As[BM][BK];   // 16 KB
  __shared__ half_t Bs[BN][BK];   // 16 KB, B^T (n-major, k contiguous)

  const int t    = threadIdx.x;
  const int wave = t >> 5;
  const int lane = t & 31;
  const int l15  = lane & 15;
  const int lh   = (lane >= 16) ? 1 : 0;
  const int wm   = wave >> 2;          // 0..1
  const int wn   = wave & 3;           // 0..3
  const int bm   = blockIdx.y * BM;
  const int bn   = blockIdx.x * BN;
  const int KT   = K / BK;

  v8f acc[4][2];
  for (int i = 0; i < 4; ++i)
    for (int j = 0; j < 2; ++j)
      acc[i][j] = (v8f){0.f, 0.f, 0.f, 0.f, 0.f, 0.f, 0.f, 0.f};

  float4 bregs[8];
  float4 aregs[8];

  auto loadB = [&](int kt) {
    for (int i = 0; i < 8; ++i) {
      int s = t + i * 256;               // 2048 float4: k(64) x n4(32)
      int kk = s >> 5, n4 = s & 31;
      bregs[i] = *(const float4*)&B[(size_t)(kt * BK + kk) * N + bn + n4 * 4];
    }
  };
  auto loadA = [&](int kt) {
    const float* A = (const float*)Av;
    for (int i = 0; i < 8; ++i) {
      int s = t + i * 256;               // 2048 float4: row(128) x c4(16)
      int row = s >> 4, c4 = s & 15;
      aregs[i] = *(const float4*)&A[(size_t)(bm + row) * K + kt * BK + c4 * 4];
    }
  };

  loadB(0);
  if (!AH) loadA(0);

  for (int kt = 0; kt < KT; ++kt) {
    if (kt) __syncthreads();             // prior tile fully consumed
    // ---- stage A tile ----
    if (AH) {
      const half_t* A = (const half_t*)Av;
      for (int i = 0; i < 2; ++i) {
        int s = t + i * 256;             // 512 uint4: row(128) x c8(4)
        int row = s >> 2, c8 = s & 3;
        async_copy_b128(lds_off(&As[row][c8 * 8]),
                        &A[(size_t)(bm + row) * K + kt * BK + c8 * 8]);
      }
    } else {
      for (int i = 0; i < 8; ++i) {
        int s = t + i * 256;
        int row = s >> 4, c4 = s & 15;
        union { half_t h[4]; uint2 u; } p;
        p.h[0] = (half_t)aregs[i].x; p.h[1] = (half_t)aregs[i].y;
        p.h[2] = (half_t)aregs[i].z; p.h[3] = (half_t)aregs[i].w;
        *(uint2*)&As[row][c4 * 4] = p.u;
      }
    }
    // ---- stage B tile: convert + transpose scatter ----
    for (int i = 0; i < 8; ++i) {
      int s = t + i * 256;
      int kk = s >> 5, n4 = s & 31;
      Bs[n4 * 4 + 0][kk] = (half_t)bregs[i].x;
      Bs[n4 * 4 + 1][kk] = (half_t)bregs[i].y;
      Bs[n4 * 4 + 2][kk] = (half_t)bregs[i].z;
      Bs[n4 * 4 + 3][kk] = (half_t)bregs[i].w;
    }
    // ---- prefetch next tile into registers (overlaps WMMA below) ----
    if (kt + 1 < KT) {
      loadB(kt + 1);
      if (!AH) loadA(kt + 1);
    }
    if (AH) wait_async0();
    __syncthreads();

    // ---- compute: 2 K-steps of 32, 8 WMMAs each ----
    for (int ks = 0; ks < 2; ++ks) {
      Frag a[4], b[2];
      int q0 = lh ? 2 : 0, q1 = q0 + 1;
      for (int mt = 0; mt < 4; ++mt) {
        int m = wm * 64 + mt * 16 + l15;
        a[mt].q[0] = *(const uint4*)&As[m][ks * 32 + q0 * 8];
        a[mt].q[1] = *(const uint4*)&As[m][ks * 32 + q1 * 8];
      }
      int kb2 = ks * 32 + (lh ? 16 : 0);
      for (int nt = 0; nt < 2; ++nt) {
        int n = wn * 32 + nt * 16 + l15;
        b[nt].q[0] = *(const uint4*)&Bs[n][kb2];
        b[nt].q[1] = *(const uint4*)&Bs[n][kb2 + 8];
      }
      for (int mt = 0; mt < 4; ++mt)
        for (int nt = 0; nt < 2; ++nt)
          acc[mt][nt] = wmma_f16(a[mt].h, b[nt].h, acc[mt][nt]);
    }
  }

  // ---- epilogue ----
  for (int mt = 0; mt < 4; ++mt)
    for (int nt = 0; nt < 2; ++nt) {
      if (CM == C_F16T) {
        // C[n][m]; 8 consecutive m per lane -> one packed 16B store
        union { half_t h[8]; uint4 u; } p;
        for (int r = 0; r < 8; ++r) p.h[r] = (half_t)acc[mt][nt][r];
        int n  = bn + wn * 32 + nt * 16 + l15;
        int m0 = bm + wm * 64 + mt * 16 + lh * 8;
        *(uint4*)&((half_t*)Cv)[(size_t)n * M + m0] = p.u;
      } else {
        for (int r = 0; r < 8; ++r) {
          int m = bm + wm * 64 + mt * 16 + r + lh * 8;
          int n = bn + wn * 32 + nt * 16 + l15;
          if (CM == C_F16)
            ((half_t*)Cv)[(size_t)m * N + n] = (half_t)acc[mt][nt][r];
          else
            ((float*)Cv)[(size_t)m * N + n] = acc[mt][nt][r];
        }
      }
    }
}

// ---------------------------------------------------------------------------
// In-place rotate-half RoPE on f16 [L, nheads, 128]; thread owns pair (i,i+64)
// ---------------------------------------------------------------------------
__global__ __launch_bounds__(256) void rope_kernel(half_t* __restrict__ qk,
                                                   int nheads,
                                                   const int* __restrict__ start) {
  size_t idx = (size_t)blockIdx.x * blockDim.x + threadIdx.x;
  int i = (int)(idx & 63);
  int h = (int)((idx >> 6) % nheads);
  int l = (int)(idx / ((size_t)64 * nheads));
  float inv = expf(-(float)i * (9.210340371976184f / 64.0f));  // 10000^(-i/64)
  float ang = (float)(l + start[0]) * inv;
  float s, c;
  sincosf(ang, &s, &c);
  half_t* p = qk + ((size_t)l * nheads + h) * 128;
  float t1 = (float)p[i];
  float t2 = (float)p[i + 64];
  p[i]      = (half_t)(t1 * c - t2 * s);
  p[i + 64] = (half_t)(t1 * s + t2 * c);
}

// ---------------------------------------------------------------------------
// Causal flash attention, GQA 32q/8kv, d=128, S=L=2048.
// grid (16 row-blocks, 32 heads); 8 waves x 16 query rows; key blocks of 64.
// K / V^T tiles double-buffered in LDS via async global->LDS copies:
// block kb+1 is prefetched while block kb is computed.
// ---------------------------------------------------------------------------
__global__ __launch_bounds__(256) void attn_kernel(const half_t* __restrict__ Qh,
                                                   const half_t* __restrict__ Kh,
                                                   const half_t* __restrict__ Vt,
                                                   half_t* __restrict__ Oh) {
  __shared__ half_t Ks[2][64][128];   // K tiles   [key][d]   32 KB
  __shared__ half_t Vs[2][128][64];   // V^T tiles [d][key]   32 KB
  __shared__ half_t Ps[8][16][64];    // per-wave P scratch   16 KB

  const int t    = threadIdx.x;
  const int wave = t >> 5;
  const int lane = t & 31;
  const int l15  = lane & 15;
  const int lh   = (lane >= 16) ? 1 : 0;
  const int rb   = blockIdx.x;
  const int head = blockIdx.y;
  const int kvh  = head >> 2;
  const int qrow0 = rb * 128 + wave * 16;

  // issue 4 async copies per thread for key-block kb into buffer buf
  auto stage = [&](int kb, int buf) {
    for (int i = 0; i < 2; ++i) {
      int s = t + i * 256;                  // 512 uint4: row(32) x c8(16)
      int row = s >> 4, c8 = s & 15;
      async_copy_b128(lds_off(&Ks[buf][row][c8 * 8]),
                      &Kh[((size_t)(kb * 64 + row) * 8 + kvh) * 128 + c8 * 8]);
    }
    for (int i = 0; i < 2; ++i) {
      int s = t + i * 256;                  // 512 uint4: d(64) x c8(8)
      int d = s >> 3, c8 = s & 7;
      async_copy_b128(lds_off(&Vs[buf][d][c8 * 8]),
                      &Vt[(size_t)(kvh * 128 + d) * 2048 + kb * 64 + c8 * 8]);
    }
  };

  // Q fragments straight from global (16B-aligned K-quarters)
  Frag qf[4];
  {
    const half_t* qp = Qh + (size_t)(qrow0 + l15) * 4096 + head * 128;
    int q0 = lh ? 2 : 0, q1 = q0 + 1;
    for (int ks = 0; ks < 4; ++ks) {
      qf[ks].q[0] = *(const uint4*)&qp[ks * 32 + q0 * 8];
      qf[ks].q[1] = *(const uint4*)&qp[ks * 32 + q1 * 8];
    }
  }

  v8f o[8];
  for (int d = 0; d < 8; ++d)
    o[d] = (v8f){0.f, 0.f, 0.f, 0.f, 0.f, 0.f, 0.f, 0.f};
  float mrow[8], lrow[8];
  for (int r = 0; r < 8; ++r) { mrow[r] = -1e30f; lrow[r] = 0.f; }

  const float scale = 0.08838834764831845f;  // 1/sqrt(128)
  const int kbmax = 2 * rb + 1;

  stage(0, 0);  // prologue prefetch

  for (int kb = 0; kb <= kbmax; ++kb) {
    const int cur = kb & 1;
    __syncthreads();   // all waves done computing on the buffer we overwrite
    if (kb < kbmax) {
      stage(kb + 1, cur ^ 1);
      wait_async4();   // 4 just issued outstanding => current buffer landed
    } else {
      wait_async0();
    }
    __syncthreads();   // buffer `cur` visible to all waves

    // waves whose rows are entirely before this key block: nothing to do
    if (kb * 64 > qrow0 + 15) continue;
    const bool needMask = (kb * 64 + 63) > qrow0;

    // ---- S = Q K^T (preload all 4 K-frags per tile for DS/WMMA overlap) ----
    v8f sacc[4];
    for (int nt = 0; nt < 4; ++nt)
      sacc[nt] = (v8f){0.f, 0.f, 0.f, 0.f, 0.f, 0.f, 0.f, 0.f};
    for (int nt = 0; nt < 4; ++nt) {
      int n = nt * 16 + l15;
      int o16 = lh ? 16 : 0;
      Frag b[4];
      for (int ks = 0; ks < 4; ++ks) {
        b[ks].q[0] = *(const uint4*)&Ks[cur][n][ks * 32 + o16];
        b[ks].q[1] = *(const uint4*)&Ks[cur][n][ks * 32 + o16 + 8];
      }
      for (int ks = 0; ks < 4; ++ks)
        sacc[nt] = wmma_f16(qf[ks].h, b[ks].h, sacc[nt]);
    }

    // ---- scale (+ causal mask on diagonal blocks only) ----
    for (int nt = 0; nt < 4; ++nt)
      for (int r = 0; r < 8; ++r) {
        float sv = sacc[nt][r] * scale;
        if (needMask) {
          int n = kb * 64 + nt * 16 + l15;
          int m = qrow0 + r + lh * 8;
          sv = (n <= m) ? sv : -1e30f;
        }
        sacc[nt][r] = sv;
      }

    // ---- online softmax (rows live within a 16-lane half) ----
    float alpha[8];
    for (int r = 0; r < 8; ++r) {
      float mx = sacc[0][r];
      for (int nt = 1; nt < 4; ++nt) mx = fmaxf(mx, sacc[nt][r]);
      for (int off = 1; off < 16; off <<= 1)
        mx = fmaxf(mx, __shfl_xor(mx, off, 32));
      float mnew = fmaxf(mrow[r], mx);
      float sum = 0.f;
      for (int nt = 0; nt < 4; ++nt) {
        float pv = __expf(sacc[nt][r] - mnew);
        sacc[nt][r] = pv;
        sum += pv;
      }
      for (int off = 1; off < 16; off <<= 1)
        sum += __shfl_xor(sum, off, 32);
      alpha[r] = __expf(mrow[r] - mnew);
      lrow[r] = lrow[r] * alpha[r] + sum;
      mrow[r] = mnew;
    }
    for (int d = 0; d < 8; ++d)
      for (int r = 0; r < 8; ++r)
        o[d][r] *= alpha[r];

    // ---- P: C-layout -> wave-private LDS -> A-layout fragments ----
    for (int nt = 0; nt < 4; ++nt)
      for (int r = 0; r < 8; ++r)
        Ps[wave][r + lh * 8][nt * 16 + l15] = (half_t)sacc[nt][r];
    wait_ds();  // wave-local LDS turnaround (DS ops are in-order per wave)

    // ---- O += P V (paired V-frag preloads for DS/WMMA overlap) ----
    for (int s2 = 0; s2 < 2; ++s2) {
      Frag pfr;
      int q0 = lh ? 2 : 0, q1 = q0 + 1;
      pfr.q[0] = *(const uint4*)&Ps[wave][l15][s2 * 32 + q0 * 8];
      pfr.q[1] = *(const uint4*)&Ps[wave][l15][s2 * 32 + q1 * 8];
      int kk = s2 * 32 + (lh ? 16 : 0);
      for (int dt = 0; dt < 8; dt += 2) {
        Frag b0, b1;
        int d0 = dt * 16 + l15;
        b0.q[0] = *(const uint4*)&Vs[cur][d0][kk];
        b0.q[1] = *(const uint4*)&Vs[cur][d0][kk + 8];
        b1.q[0] = *(const uint4*)&Vs[cur][d0 + 16][kk];
        b1.q[1] = *(const uint4*)&Vs[cur][d0 + 16][kk + 8];
        o[dt]     = wmma_f16(pfr.h, b0.h, o[dt]);
        o[dt + 1] = wmma_f16(pfr.h, b1.h, o[dt + 1]);
      }
    }
  }

  // ---- normalize, write f16 attn output [L, 32*128] ----
  for (int dt = 0; dt < 8; ++dt)
    for (int r = 0; r < 8; ++r) {
      int m = qrow0 + r + lh * 8;
      int d = dt * 16 + l15;
      Oh[(size_t)m * 4096 + head * 128 + d] = (half_t)(o[dt][r] / lrow[r]);
    }
}

// ---------------------------------------------------------------------------
extern "C" void kernel_launch(void* const* d_in, const int* in_sizes, int n_in,
                              void* d_out, int out_size, void* d_ws, size_t ws_size,
                              hipStream_t stream) {
  const float* x  = (const float*)d_in[0];
  const float* Wq = (const float*)d_in[1];
  const float* Wk = (const float*)d_in[2];
  const float* Wv = (const float*)d_in[3];
  const float* Wo = (const float*)d_in[4];
  const int* start = (const int*)d_in[5];

  const size_t L = 2048;
  half_t* Qh = (half_t*)d_ws;                 // [L, 4096]
  half_t* Kh = Qh + L * 4096;                 // [L, 8, 128]
  half_t* Vt = Kh + L * 1024;                 // [8*128, L]  (transposed V)
  half_t* Ah = Vt + L * 1024;                 // [L, 4096]

  dim3 blk(256);
  gemm_kernel<false, C_F16><<<dim3(4096 / BN, 2048 / BM), blk, 0, stream>>>(
      x, Wq, Qh, 2048, 4096, 4096);
  gemm_kernel<false, C_F16><<<dim3(1024 / BN, 2048 / BM), blk, 0, stream>>>(
      x, Wk, Kh, 2048, 1024, 4096);
  gemm_kernel<false, C_F16T><<<dim3(1024 / BN, 2048 / BM), blk, 0, stream>>>(
      x, Wv, Vt, 2048, 1024, 4096);
  rope_kernel<<<(2048 * 32 * 64) / 256, 256, 0, stream>>>(Qh, 32, start);
  rope_kernel<<<(2048 * 8 * 64) / 256, 256, 0, stream>>>(Kh, 8, start);
  attn_kernel<<<dim3(16, 32), blk, 0, stream>>>(Qh, Kh, Vt, Ah);
  gemm_kernel<true, C_F32><<<dim3(4096 / BN, 2048 / BM), blk, 0, stream>>>(
      Ah, Wo, d_out, 2048, 4096, 4096);
}
